// GE_PerSectionPredNet_71674414235712
// MI455X (gfx1250) — compile-verified, matrix operand order checked by reading, not run on hardware
//
#include <hip/hip_runtime.h>
#include <hip/hip_bf16.h>
#include <math.h>

// ---------------------------------------------------------------------------
// Problem constants (from reference)
// ---------------------------------------------------------------------------
#define NN    10000   // nodes
#define EE    320000  // edges
#define F_IN  8192
#define HH    200
#define HID   400
#define MSG_EPS 1e-7f
#define LN_EPS  1e-5f

typedef __attribute__((ext_vector_type(16))) __bf16 v16bf;
typedef __attribute__((ext_vector_type(8)))  float  v8f;

// ---------------------------------------------------------------------------
// WMMA 16x32 bf16 fragment from an fp32 row-major matrix.
// ISA layout (CDNA5 ISA, 16-bit A 16x32):
//   lanes 0-15  : row = lane,    VGPR0..3 -> K 0..7,   VGPR4..7 -> K 16..23
//   lanes 16-31 : row = lane-16, VGPR0..3 -> K 8..15,  VGPR4..7 -> K 24..31
// rp = per-lane row pointer (row pre-clamped -> loads are UNCONDITIONAL).
// ---------------------------------------------------------------------------
__device__ __forceinline__ v16bf load_frag_full(const float* __restrict__ rp,
                                                int k0, int khalf) {
  v16bf f;
#pragma unroll
  for (int v = 0; v < 8; ++v) {
    const int kb = (v < 4 ? 2 * v : 16 + 2 * (v - 4)) + khalf * 8;
    const float2 d = *(const float2*)(rp + k0 + kb);   // even k, 8B aligned
    f[2 * v]     = (__bf16)d.x;
    f[2 * v + 1] = (__bf16)d.y;
  }
  return f;
}

// Tail variant: k may exceed K. Clamp the address (row pointer stays valid),
// zero the lanes-out-of-K via a VALU mask -> still no exec-mask branches.
__device__ __forceinline__ v16bf load_frag_tail(const float* __restrict__ rp,
                                                int k0, int khalf, int K) {
  v16bf f;
#pragma unroll
  for (int v = 0; v < 8; ++v) {
    const int kb = (v < 4 ? 2 * v : 16 + 2 * (v - 4)) + khalf * 8;
    const int k  = k0 + kb;
    const int kc = (k < K) ? k : 0;          // clamped address, always in-bounds
    const float m = (k < K) ? 1.f : 0.f;
    const float2 d = *(const float2*)(rp + kc);
    f[2 * v]     = (__bf16)(d.x * m);
    f[2 * v + 1] = (__bf16)(d.y * m);
  }
  return f;
}

// ---------------------------------------------------------------------------
// C = act(A[M,K] @ W[Nc,K]^T + bias), bf16 WMMA, f32 accumulate.
// Block = 128 threads (4 waves). Each wave: 32 rows x 64 cols = 8 WMMAs/K-step
// (2 A-frags, 4 B-frags). Block tile = 128 x 64.  ACT: 0 = none, 1 = relu.
// ---------------------------------------------------------------------------
template <int ACT>
__global__ __launch_bounds__(128)
void wmma_gemm_kernel(const float* __restrict__ A, const float* __restrict__ W,
                      const float* __restrict__ bias, float* __restrict__ C,
                      int M, int Nc, int K) {
  const int lane  = threadIdx.x & 31;
  const int wave  = threadIdx.x >> 5;               // 0..3
  const int r0    = blockIdx.x * 128 + wave * 32;
  const int c0    = blockIdx.y * 64;
  const int khalf = lane >> 4;
  const int lrow  = lane & 15;

  // Pre-clamped per-lane row pointers -> all loads unconditional.
  const float* pa0 = A + (size_t)min(r0 + lrow,      M - 1)  * (size_t)K;
  const float* pa1 = A + (size_t)min(r0 + 16 + lrow, M - 1)  * (size_t)K;
  const float* pb0 = W + (size_t)min(c0 +  0 + lrow, Nc - 1) * (size_t)K;
  const float* pb1 = W + (size_t)min(c0 + 16 + lrow, Nc - 1) * (size_t)K;
  const float* pb2 = W + (size_t)min(c0 + 32 + lrow, Nc - 1) * (size_t)K;
  const float* pb3 = W + (size_t)min(c0 + 48 + lrow, Nc - 1) * (size_t)K;

  v8f acc[2][4];
#pragma unroll
  for (int i = 0; i < 2; ++i)
#pragma unroll
    for (int j = 0; j < 4; ++j) acc[i][j] = (v8f){};

  const int KFULL = K & ~31;
  int k0 = 0;
  for (; k0 < KFULL; k0 += 32) {
    if (k0 + 32 < KFULL) {
      __builtin_prefetch(pa0 + k0 + 32, 0, 1);      // global_prefetch next K tile
      __builtin_prefetch(pa1 + k0 + 32, 0, 1);
    }
    const v16bf a0 = load_frag_full(pa0, k0, khalf);
    const v16bf a1 = load_frag_full(pa1, k0, khalf);
    const v16bf b0 = load_frag_full(pb0, k0, khalf);
    const v16bf b1 = load_frag_full(pb1, k0, khalf);
    const v16bf b2 = load_frag_full(pb2, k0, khalf);
    const v16bf b3 = load_frag_full(pb3, k0, khalf);
    acc[0][0] = __builtin_amdgcn_wmma_f32_16x16x32_bf16(false, a0, false, b0, (short)0, acc[0][0], false, false);
    acc[0][1] = __builtin_amdgcn_wmma_f32_16x16x32_bf16(false, a0, false, b1, (short)0, acc[0][1], false, false);
    acc[0][2] = __builtin_amdgcn_wmma_f32_16x16x32_bf16(false, a0, false, b2, (short)0, acc[0][2], false, false);
    acc[0][3] = __builtin_amdgcn_wmma_f32_16x16x32_bf16(false, a0, false, b3, (short)0, acc[0][3], false, false);
    acc[1][0] = __builtin_amdgcn_wmma_f32_16x16x32_bf16(false, a1, false, b0, (short)0, acc[1][0], false, false);
    acc[1][1] = __builtin_amdgcn_wmma_f32_16x16x32_bf16(false, a1, false, b1, (short)0, acc[1][1], false, false);
    acc[1][2] = __builtin_amdgcn_wmma_f32_16x16x32_bf16(false, a1, false, b2, (short)0, acc[1][2], false, false);
    acc[1][3] = __builtin_amdgcn_wmma_f32_16x16x32_bf16(false, a1, false, b3, (short)0, acc[1][3], false, false);
  }
  if (k0 < K) {  // single masked tail iteration (K=200 or 400 cases)
    const v16bf a0 = load_frag_tail(pa0, k0, khalf, K);
    const v16bf a1 = load_frag_tail(pa1, k0, khalf, K);
    const v16bf b0 = load_frag_tail(pb0, k0, khalf, K);
    const v16bf b1 = load_frag_tail(pb1, k0, khalf, K);
    const v16bf b2 = load_frag_tail(pb2, k0, khalf, K);
    const v16bf b3 = load_frag_tail(pb3, k0, khalf, K);
    acc[0][0] = __builtin_amdgcn_wmma_f32_16x16x32_bf16(false, a0, false, b0, (short)0, acc[0][0], false, false);
    acc[0][1] = __builtin_amdgcn_wmma_f32_16x16x32_bf16(false, a0, false, b1, (short)0, acc[0][1], false, false);
    acc[0][2] = __builtin_amdgcn_wmma_f32_16x16x32_bf16(false, a0, false, b2, (short)0, acc[0][2], false, false);
    acc[0][3] = __builtin_amdgcn_wmma_f32_16x16x32_bf16(false, a0, false, b3, (short)0, acc[0][3], false, false);
    acc[1][0] = __builtin_amdgcn_wmma_f32_16x16x32_bf16(false, a1, false, b0, (short)0, acc[1][0], false, false);
    acc[1][1] = __builtin_amdgcn_wmma_f32_16x16x32_bf16(false, a1, false, b1, (short)0, acc[1][1], false, false);
    acc[1][2] = __builtin_amdgcn_wmma_f32_16x16x32_bf16(false, a1, false, b2, (short)0, acc[1][2], false, false);
    acc[1][3] = __builtin_amdgcn_wmma_f32_16x16x32_bf16(false, a1, false, b3, (short)0, acc[1][3], false, false);
  }

  // C/D layout: VGPR r -> row (lane<16 ? r : r+8), col = lane&15
  const int colb   = c0 + lrow;
  const int rowoff = r0 + (khalf << 3);
#pragma unroll
  for (int i = 0; i < 2; ++i) {
#pragma unroll
    for (int j = 0; j < 4; ++j) {
      const int col = colb + 16 * j;
      if (col >= Nc) continue;
      const float bv = bias ? bias[col] : 0.f;
#pragma unroll
      for (int r = 0; r < 8; ++r) {
        const int row = rowoff + 16 * i + r;
        if (row < M) {
          float v = acc[i][j][r] + bv;
          if (ACT == 1) v = fmaxf(v, 0.f);
          C[(size_t)row * Nc + col] = v;
        }
      }
    }
  }
}

// ---------------------------------------------------------------------------
// Elementwise zero
// ---------------------------------------------------------------------------
__global__ void zero_kernel(float* __restrict__ p, long long n) {
  long long i = (long long)blockIdx.x * blockDim.x + threadIdx.x;
  long long stride = (long long)gridDim.x * blockDim.x;
  for (; i < n; i += stride) p[i] = 0.f;
}

// ---------------------------------------------------------------------------
// Edge pass 1: per-(dst,channel) running max of st = (relu(x[src])+eps)*t
// st > 0 for t > 0 -> uint-ordered atomicMax vs 0-initialized buffer matches
// float ordering. Warp-per-edge -> coalesced 800B row gather.
// ---------------------------------------------------------------------------
__global__ __launch_bounds__(256)
void edge_max_kernel(const float* __restrict__ x, const int* __restrict__ src,
                     const int* __restrict__ dst, const float* __restrict__ t,
                     int layer, unsigned* __restrict__ mmax) {
  const int e    = (int)((blockIdx.x * blockDim.x + threadIdx.x) >> 5);
  const int lane = threadIdx.x & 31;
  if (e >= EE) return;
  const float tv = t[layer];
  const int s = src[e], d = dst[e];
  const float* xs = x + (size_t)s * HH;
  unsigned* md = mmax + (size_t)d * HH;
  for (int c = lane; c < HH; c += 32) {
    const float st = (fmaxf(xs[c], 0.f) + MSG_EPS) * tv;
    atomicMax(&md[c], __float_as_uint(st));
  }
}

// ---------------------------------------------------------------------------
// Edge pass 2 (fused softmax): numer += msg*exp(st-m), denom += exp(st-m)
// agg = numer/denom == sum(msg * softmax(st))  (algebraically identical)
// ---------------------------------------------------------------------------
__global__ __launch_bounds__(256)
void edge_sum_kernel(const float* __restrict__ x, const int* __restrict__ src,
                     const int* __restrict__ dst, const float* __restrict__ t,
                     int layer, const unsigned* __restrict__ mmax,
                     float* __restrict__ numer, float* __restrict__ denom) {
  const int e    = (int)((blockIdx.x * blockDim.x + threadIdx.x) >> 5);
  const int lane = threadIdx.x & 31;
  if (e >= EE) return;
  const float tv = t[layer];
  const int s = src[e], d = dst[e];
  const float* xs = x + (size_t)s * HH;
  const unsigned* md = mmax + (size_t)d * HH;
  float* np = numer + (size_t)d * HH;
  float* dp = denom + (size_t)d * HH;
  for (int c = lane; c < HH; c += 32) {
    const float msg = fmaxf(xs[c], 0.f) + MSG_EPS;
    const float st  = msg * tv;
    const float ex  = __expf(st - __uint_as_float(md[c]));
    atomicAdd(&np[c], msg * ex);
    atomicAdd(&dp[c], ex);
  }
}

// ---------------------------------------------------------------------------
// y = agg + x  (residual of x_r inside GENConv)
// ---------------------------------------------------------------------------
__global__ void combine_kernel(const float* __restrict__ x,
                               const float* __restrict__ numer,
                               const float* __restrict__ denom,
                               float* __restrict__ y, long long n) {
  long long i = (long long)blockIdx.x * blockDim.x + threadIdx.x;
  long long stride = (long long)gridDim.x * blockDim.x;
  for (; i < n; i += stride) {
    const float dn = denom[i];
    const float agg = dn > 0.f ? numer[i] / dn : 0.f;
    y[i] = agg + x[i];
  }
}

// ---------------------------------------------------------------------------
// Wave-per-row LayerNorm + ReLU (in place)
// ---------------------------------------------------------------------------
__global__ __launch_bounds__(256)
void ln_relu_kernel(float* __restrict__ h, const float* __restrict__ g,
                    const float* __restrict__ b, int n, int L) {
  const int row  = (int)((blockIdx.x * blockDim.x + threadIdx.x) >> 5);
  const int lane = threadIdx.x & 31;
  if (row >= n) return;
  float* hp = h + (size_t)row * L;
  float s = 0.f, s2 = 0.f;
  for (int c = lane; c < L; c += 32) { const float v = hp[c]; s += v; s2 += v * v; }
#pragma unroll
  for (int o = 16; o > 0; o >>= 1) { s += __shfl_xor(s, o, 32); s2 += __shfl_xor(s2, o, 32); }
  const float mu = s / (float)L;
  const float var = s2 / (float)L - mu * mu;
  const float rs = rsqrtf(var + LN_EPS);
  for (int c = lane; c < L; c += 32) {
    const float v = (hp[c] - mu) * rs * g[c] + b[c];
    hp[c] = fmaxf(v, 0.f);
  }
}

// ---------------------------------------------------------------------------
// x += leaky_relu(LN(h), 0.2)   (DeepGCNLayer 'res' block tail)
// ---------------------------------------------------------------------------
__global__ __launch_bounds__(256)
void ln_leaky_res_kernel(float* __restrict__ x, const float* __restrict__ h,
                         const float* __restrict__ g, const float* __restrict__ b,
                         int n, int L) {
  const int row  = (int)((blockIdx.x * blockDim.x + threadIdx.x) >> 5);
  const int lane = threadIdx.x & 31;
  if (row >= n) return;
  const float* hp = h + (size_t)row * L;
  float* xp = x + (size_t)row * L;
  float s = 0.f, s2 = 0.f;
  for (int c = lane; c < L; c += 32) { const float v = hp[c]; s += v; s2 += v * v; }
#pragma unroll
  for (int o = 16; o > 0; o >>= 1) { s += __shfl_xor(s, o, 32); s2 += __shfl_xor(s2, o, 32); }
  const float mu = s / (float)L;
  const float var = s2 / (float)L - mu * mu;
  const float rs = rsqrtf(var + LN_EPS);
  for (int c = lane; c < L; c += 32) {
    const float v = (hp[c] - mu) * rs * g[c] + b[c];
    xp[c] += (v > 0.f ? v : 0.2f * v);
  }
}

// ---------------------------------------------------------------------------
// out[i] = sigmoid(dot(x[i,:], W_out) + b_out), wave per node
// ---------------------------------------------------------------------------
__global__ __launch_bounds__(256)
void out_kernel(const float* __restrict__ x, const float* __restrict__ wout,
                const float* __restrict__ bout, float* __restrict__ out, int n) {
  const int row  = (int)((blockIdx.x * blockDim.x + threadIdx.x) >> 5);
  const int lane = threadIdx.x & 31;
  if (row >= n) return;
  const float* xp = x + (size_t)row * HH;
  float s = 0.f;
  for (int c = lane; c < HH; c += 32) s += xp[c] * wout[c];
#pragma unroll
  for (int o = 16; o > 0; o >>= 1) s += __shfl_xor(s, o, 32);
  if (lane == 0) out[row] = 1.f / (1.f + expf(-(s + bout[0])));
}

// ---------------------------------------------------------------------------
// Host-side orchestration (graph-capture safe: only kernel launches on stream)
// ---------------------------------------------------------------------------
extern "C" void kernel_launch(void* const* d_in, const int* in_sizes, int n_in,
                              void* d_out, int out_size, void* d_ws, size_t ws_size,
                              hipStream_t stream) {
  // inputs in setup_inputs() order
  const float* PPI_x    = (const float*)d_in[0];
  const int*   edge_idx = (const int*)d_in[1];     // [2,E]: src then dst
  const float* W_in     = (const float*)d_in[4];
  const float* b_in     = (const float*)d_in[5];
  const float* t        = (const float*)d_in[6];
  const float* W1       = (const float*)d_in[7];   // [3,HID,H]
  const float* b1       = (const float*)d_in[8];   // [3,HID]
  const float* g1       = (const float*)d_in[9];
  const float* be1      = (const float*)d_in[10];
  const float* W2       = (const float*)d_in[11];  // [3,H,HID]
  const float* b2       = (const float*)d_in[12];
  const float* go       = (const float*)d_in[13];
  const float* bo       = (const float*)d_in[14];
  const float* W_out    = (const float*)d_in[15];
  const float* b_out    = (const float*)d_in[16];
  float* out = (float*)d_out;

  const int* src = edge_idx;
  const int* dst = edge_idx + EE;

  // workspace layout (floats), ~56 MB total
  float* ws = (float*)d_ws;
  float* x     = ws;                       // [N,H]
  float* y     = x     + (size_t)NN * HH;  // [N,H]
  float* h1    = y     + (size_t)NN * HH;  // [N,HID]
  float* mmaxf = h1    + (size_t)NN * HID; // [N,H]
  float* numer = mmaxf + (size_t)NN * HH;  // [N,H]
  float* denom = numer + (size_t)NN * HH;  // [N,H]

  const dim3 gblk(128);
  const dim3 ggrid_in((NN + 127) / 128, (HH + 63) / 64);   // 79 x 4
  const dim3 ggrid_m1((NN + 127) / 128, (HID + 63) / 64);  // 79 x 7
  const dim3 ggrid_m2((NN + 127) / 128, (HH + 63) / 64);

  const int ethreads = 256;
  const int eblocks  = (EE + 7) / 8;       // warp per edge
  const int rblocks  = (NN + 7) / 8;       // wave per node row
  const long long nh = (long long)NN * HH;

  // 1) x = relu(PPI_x @ W_in^T + b_in)    [dominant GEMM, bf16 WMMA]
  wmma_gemm_kernel<1><<<ggrid_in, gblk, 0, stream>>>(PPI_x, W_in, b_in, x, NN, HH, F_IN);

  // 2) three DeepGCN layers
  for (int l = 0; l < 3; ++l) {
    zero_kernel<<<2048, 256, 0, stream>>>(mmaxf, 3 * nh);  // mmax|numer|denom contiguous
    edge_max_kernel<<<eblocks, ethreads, 0, stream>>>(x, src, dst, t, l, (unsigned*)mmaxf);
    edge_sum_kernel<<<eblocks, ethreads, 0, stream>>>(x, src, dst, t, l,
                                                      (const unsigned*)mmaxf, numer, denom);
    combine_kernel<<<2048, 256, 0, stream>>>(x, numer, denom, y, nh);

    // GENConv MLP: Lin(H->HID) + LN + ReLU + Lin(HID->H)
    wmma_gemm_kernel<0><<<ggrid_m1, gblk, 0, stream>>>(y, W1 + (size_t)l * HID * HH,
                                                       b1 + (size_t)l * HID, h1, NN, HID, HH);
    ln_relu_kernel<<<rblocks, 256, 0, stream>>>(h1, g1 + (size_t)l * HID,
                                                be1 + (size_t)l * HID, NN, HID);
    wmma_gemm_kernel<0><<<ggrid_m2, gblk, 0, stream>>>(h1, W2 + (size_t)l * HH * HID,
                                                       b2 + (size_t)l * HH, y, NN, HH, HID);
    // x += leaky_relu(LN(h), 0.2)
    ln_leaky_res_kernel<<<rblocks, 256, 0, stream>>>(x, y, go + (size_t)l * HH,
                                                     bo + (size_t)l * HH, NN, HH);
  }

  // 3) out = sigmoid(x @ W_out^T + b_out)
  out_kernel<<<rblocks, 256, 0, stream>>>(x, W_out, b_out, out, NN);
}
